// WaveKANFFN_81853486727585
// MI455X (gfx1250) — matrix operand order repaired and checked by reading, not compiled, not run on hardware
//
#include <hip/hip_runtime.h>

// ---------------------------------------------------------------------------
// WaveKAN FFN for MI455X (gfx1250, wave32, WMMA).
// Fused dual-GEMM per layer (K' = 2K, [wav|silu] packed bf16 activations x
// [W_wav|W_base] packed bf16 weights), f32 accum via v_wmma_f32_16x16x32_bf16.
// Wave tile 64x64 (16 WMMAs / K-step), block tile 128x256, double-buffered
// LDS fed by GLOBAL_LOAD_ASYNC_TO_LDS_B128 (ASYNCcnt) when available.
// ---------------------------------------------------------------------------

typedef __attribute__((ext_vector_type(8)))  float        v8f;
typedef __attribute__((ext_vector_type(16))) __bf16       v16bf;
typedef __attribute__((ext_vector_type(4)))  unsigned int v4u;
typedef __attribute__((ext_vector_type(4)))  int          v4i;

#if defined(__has_builtin)
#  if __has_builtin(__builtin_amdgcn_global_load_async_to_lds_b128)
#    define HAVE_ASYNC_LDS 1
#  endif
#  if __has_builtin(__builtin_amdgcn_s_wait_asynccnt)
#    define HAVE_WAIT_ASYNC 1
#  endif
#endif

#define MH_C 0.8673250705840776f   // 2/sqrt(3) * pi^(-1/4)

__device__ __forceinline__ unsigned short f2bf(float f) {
    unsigned int u = __float_as_uint(f);
    u += 0x7FFFu + ((u >> 16) & 1u);     // round-to-nearest-even
    return (unsigned short)(u >> 16);
}
__device__ __forceinline__ float mexhat(float z) {
    float z2 = z * z;
    return MH_C * (1.0f - z2) * __expf(-0.5f * z2);
}
__device__ __forceinline__ float silu_f(float x) {
    return x / (1.0f + __expf(-x));
}

#if defined(HAVE_ASYNC_LDS)
typedef __attribute__((address_space(1))) v4i* gv4i_p;   // global int4*
typedef __attribute__((address_space(3))) v4i* lv4i_p;   // LDS int4*

__device__ __forceinline__ void async_b128(const unsigned short* g, unsigned short* l) {
    __builtin_amdgcn_global_load_async_to_lds_b128(
        (gv4i_p)(v4i*)const_cast<unsigned short*>(g),
        (lv4i_p)(v4i*)l, 0, 0);
}
__device__ __forceinline__ void wait_async0() {
#if defined(HAVE_WAIT_ASYNC)
    __builtin_amdgcn_s_wait_asynccnt(0);
#else
    asm volatile("s_wait_asynccnt 0x0" ::: "memory");
#endif
}
#endif

// ---------------------------------------------------------------------------
// Elementwise prep: x[M,D] -> packed bf16 activations act[M, 2D]
// ---------------------------------------------------------------------------
__global__ void wavekan_act(const float* __restrict__ x,
                            const float* __restrict__ scale,
                            const float* __restrict__ trans,
                            unsigned short* __restrict__ act,
                            long long M, int D) {
    long long idx = (long long)blockIdx.x * blockDim.x + threadIdx.x;
    long long total = M * 2LL * D;
    if (idx >= total) return;
    int K2 = 2 * D;
    long long m = idx / K2;
    int k = (int)(idx - m * (long long)K2);
    float v;
    if (k < D) {
        float xv = x[m * D + k];
        float z  = (xv - trans[k]) / scale[k];
        v = mexhat(z);
    } else {
        v = silu_f(x[m * D + (k - D)]);
    }
    act[idx] = f2bf(v);
}

// ---------------------------------------------------------------------------
// Weight pack: wav_w[O,I], base_w[O,I] (f32) -> wp[O, 2I] bf16 = [wav | base]
// ---------------------------------------------------------------------------
__global__ void wavekan_pack_w(const float* __restrict__ wav_w,
                               const float* __restrict__ base_w,
                               unsigned short* __restrict__ wp,
                               int O, int I) {
    long long idx = (long long)blockIdx.x * blockDim.x + threadIdx.x;
    long long total = (long long)O * 2 * I;
    if (idx >= total) return;
    int K2 = 2 * I;
    long long o = idx / K2;
    int k = (int)(idx - o * (long long)K2);
    float v = (k < I) ? wav_w[o * I + k] : base_w[o * I + (k - I)];
    wp[idx] = f2bf(v);
}

// ---------------------------------------------------------------------------
// NT GEMM: C[M,N] = A[M,K] * B[N,K]^T   (bf16 in, f32 accum, WMMA)
// Block tile 128x256, BK=32; 8 waves as 2(M) x 4(N); wave tile 64x64 = 16 WMMA.
// mode 0: epilogue -> packed bf16 activations actOut[M, 2N] via scale/trans
// mode 1: epilogue -> f32 outF32[M, N]
// ---------------------------------------------------------------------------
#define BM 128
#define BN 256
#define BK 32
#define LDT 40   // LDS row stride in bf16 elements (32 + 8 pad, 16B aligned)

__global__ __launch_bounds__(256)
void wavekan_gemm(const unsigned short* __restrict__ A,
                  const unsigned short* __restrict__ B,
                  long long M, int N, int K,
                  float* __restrict__ outF32,
                  unsigned short* __restrict__ actOut,
                  const float* __restrict__ scaleN,
                  const float* __restrict__ transN,
                  int mode) {
    __shared__ __align__(16) unsigned short ldsA[2 * BM * LDT];   // 20 KB
    __shared__ __align__(16) unsigned short ldsB[2 * BN * LDT];   // 40 KB

    const int tid    = threadIdx.x;
    const int lane   = tid & 31;
    const int wave   = tid >> 5;
    const int waveM  = wave >> 2;    // 0..1  (64-row half)
    const int waveN  = wave & 3;     // 0..3  (64-col quarter)
    const int lane16 = lane & 15;
    const int hi     = lane >> 4;    // 0 or 1

    const long long mBase = (long long)blockIdx.y * BM;
    const long long nBase = (long long)blockIdx.x * BN;

    v8f acc[4][4];
    for (int i = 0; i < 4; ++i)
        for (int j = 0; j < 4; ++j)
            for (int r = 0; r < 8; ++r) acc[i][j][r] = 0.0f;

    // 16 WMMAs from one LDS stage: 4 A-frags + 4 B-frags.
    auto do_mmas = [&](const unsigned short* As, const unsigned short* Bs) {
        v16bf af[4], bfm[4];
        for (int mt = 0; mt < 4; ++mt) {
            // A frag: lane<16 -> M=lane16, K {0..7,16..23}; lane>=16 -> K {8..15,24..31}
            const unsigned short* p = As + (waveM * 64 + mt * 16 + lane16) * LDT + hi * 8;
            union { v16bf v; v4u u[2]; } t;
            t.u[0] = *(const v4u*)p;
            t.u[1] = *(const v4u*)(p + 16);
            af[mt] = t.v;
        }
        for (int nt = 0; nt < 4; ++nt) {
            // B frag: lane<16 -> N=lane16, K {0..7,16..23}; lane>=16 -> K {8..15,24..31}
            const unsigned short* p = Bs + (waveN * 64 + nt * 16 + lane16) * LDT + hi * 8;
            union { v16bf v; v4u u[2]; } t;
            t.u[0] = *(const v4u*)p;
            t.u[1] = *(const v4u*)(p + 16);
            bfm[nt] = t.v;
        }
        for (int mt = 0; mt < 4; ++mt)
            for (int nt = 0; nt < 4; ++nt)
                acc[mt][nt] = __builtin_amdgcn_wmma_f32_16x16x32_bf16(
                    false, af[mt], false, bfm[nt],
                    (short)0, acc[mt][nt], false, false);
    };

#if defined(HAVE_ASYNC_LDS)
    // Async DMA path: memory -> LDS, no VGPR staging, one barrier per K-step.
    auto issue_stage = [&](unsigned short* Abuf, unsigned short* Bbuf, int k0) {
        // A tile: 128 rows x 64B = 512 16B-chunks -> 2 per thread
        for (int i = 0; i < 2; ++i) {
            int c = tid + i * 256, row = c >> 2, cc = c & 3;
            async_b128(A + (mBase + row) * (long long)K + k0 + cc * 8,
                       Abuf + row * LDT + cc * 8);
        }
        // B tile: 256 rows x 64B = 1024 chunks -> 4 per thread
        for (int i = 0; i < 4; ++i) {
            int c = tid + i * 256, row = c >> 2, cc = c & 3;
            async_b128(B + (nBase + row) * (long long)K + k0 + cc * 8,
                       Bbuf + row * LDT + cc * 8);
        }
    };

    issue_stage(ldsA, ldsB, 0);
    int stage = 0;
    for (int k0 = 0; k0 < K; k0 += BK) {
        wait_async0();          // my DMA done
        __syncthreads();        // everyone's DMA done; prior reads of other buf done
        if (k0 + BK < K)        // start next stage into the idle buffer
            issue_stage(ldsA + (stage ^ 1) * (BM * LDT),
                        ldsB + (stage ^ 1) * (BN * LDT), k0 + BK);
        do_mmas(ldsA + stage * (BM * LDT), ldsB + stage * (BN * LDT));
        stage ^= 1;
    }
#else
    // Fallback: global -> regs -> LDS (single buffer, two barriers).
    auto gload = [&](const unsigned short* base, long long rowBase, int k0, int c) -> v4u {
        int row = c >> 2, cc = c & 3;
        return *(const v4u*)(base + (rowBase + row) * (long long)K + k0 + cc * 8);
    };
    v4u ra[2], rb[4];
    for (int i = 0; i < 2; ++i) ra[i] = gload(A, mBase, 0, tid + i * 256);
    for (int i = 0; i < 4; ++i) rb[i] = gload(B, nBase, 0, tid + i * 256);

    for (int k0 = 0; k0 < K; k0 += BK) {
        __syncthreads();
        for (int i = 0; i < 2; ++i) {
            int c = tid + i * 256, row = c >> 2, cc = c & 3;
            *(v4u*)(ldsA + row * LDT + cc * 8) = ra[i];
        }
        for (int i = 0; i < 4; ++i) {
            int c = tid + i * 256, row = c >> 2, cc = c & 3;
            *(v4u*)(ldsB + row * LDT + cc * 8) = rb[i];
        }
        __syncthreads();
        if (k0 + BK < K) {
            for (int i = 0; i < 2; ++i) ra[i] = gload(A, mBase, k0 + BK, tid + i * 256);
            for (int i = 0; i < 4; ++i) rb[i] = gload(B, nBase, k0 + BK, tid + i * 256);
        }
        do_mmas(ldsA, ldsB);
    }
#endif

    // Epilogue. C/D layout: VGPR r -> M = r (+8 for lanes 16..31), N = lane16.
    if (mode == 1) {
        for (int mt = 0; mt < 4; ++mt)
            for (int nt = 0; nt < 4; ++nt) {
                long long col  = nBase + waveN * 64 + nt * 16 + lane16;
                long long rowB = mBase + waveM * 64 + mt * 16 + hi * 8;
                for (int r = 0; r < 8; ++r)
                    outF32[(rowB + r) * N + col] = acc[mt][nt][r];
            }
    } else {
        for (int mt = 0; mt < 4; ++mt)
            for (int nt = 0; nt < 4; ++nt) {
                long long col = nBase + waveN * 64 + nt * 16 + lane16;
                float rs = 1.0f / scaleN[col];
                float tr = transN[col];
                long long rowB = mBase + waveM * 64 + mt * 16 + hi * 8;
                for (int r = 0; r < 8; ++r) {
                    float h = acc[mt][nt][r];
                    long long base = (rowB + r) * (2LL * N);
                    actOut[base + col]     = f2bf(mexhat((h - tr) * rs));
                    actOut[base + N + col] = f2bf(silu_f(h));
                }
            }
    }
}

// ---------------------------------------------------------------------------
extern "C" void kernel_launch(void* const* d_in, const int* in_sizes, int n_in,
                              void* d_out, int out_size, void* d_ws, size_t ws_size,
                              hipStream_t stream) {
    const float* x       = (const float*)d_in[0];
    const float* scale1  = (const float*)d_in[1];
    const float* trans1  = (const float*)d_in[2];
    const float* wav_w1  = (const float*)d_in[3];
    const float* base_w1 = (const float*)d_in[4];
    const float* scale2  = (const float*)d_in[5];
    const float* trans2  = (const float*)d_in[6];
    const float* wav_w2  = (const float*)d_in[7];
    const float* base_w2 = (const float*)d_in[8];

    const int D = in_sizes[1];                          // 1024
    const int H = in_sizes[3] / D;                      // 4096
    const long long M = (long long)in_sizes[0] / D;     // B*S = 8192

    unsigned short* act1 = (unsigned short*)d_ws;           // [M, 2D]
    unsigned short* w1p  = act1 + M * 2LL * D;              // [H, 2D]
    unsigned short* act2 = w1p + (long long)H * 2 * D;      // [M, 2H]
    unsigned short* w2p  = act2 + M * 2LL * H;              // [D, 2H]
    float* out = (float*)d_out;                             // [M, D] f32

    {   // layer-1 activations
        long long tot = M * 2LL * D;
        wavekan_act<<<dim3((unsigned)((tot + 255) / 256)), 256, 0, stream>>>(
            x, scale1, trans1, act1, M, D);
    }
    {   // packed weights
        long long t1 = (long long)H * 2 * D;
        wavekan_pack_w<<<dim3((unsigned)((t1 + 255) / 256)), 256, 0, stream>>>(
            wav_w1, base_w1, w1p, H, D);
        long long t2 = (long long)D * 2 * H;
        wavekan_pack_w<<<dim3((unsigned)((t2 + 255) / 256)), 256, 0, stream>>>(
            wav_w2, base_w2, w2p, D, H);
    }

    // GEMM 1: [M, 2D] x [H, 2D]^T -> layer-2 packed activations [M, 2H]
    wavekan_gemm<<<dim3(H / BN, (unsigned)(M / BM)), 256, 0, stream>>>(
        act1, w1p, M, H, 2 * D, nullptr, act2, scale2, trans2, 0);

    // GEMM 2: [M, 2H] x [D, 2H]^T -> f32 output [M, D]
    wavekan_gemm<<<dim3(D / BN, (unsigned)(M / BM)), 256, 0, stream>>>(
        act2, w2p, M, D, 2 * H, out, nullptr, nullptr, nullptr, 1);
}
